// Seq2SeqAttentionModel_88381837017852
// MI455X (gfx1250) — matrix operand order repaired and checked by reading, not compiled, not run on hardware
//
#include <hip/hip_runtime.h>
#include <hip/hip_bf16.h>
#include <math.h>

// Problem dimensions (fixed by the reference).
#define B_  32
#define S_  64
#define T_  48
#define E_  512
#define H_  1024
#define V_  32000
#define TD  47   // T-1 decoder steps

typedef __attribute__((ext_vector_type(16))) __bf16       v16bf;
typedef __attribute__((ext_vector_type(8)))  unsigned int v8u;
typedef __attribute__((ext_vector_type(8)))  float        v8f;
typedef __attribute__((ext_vector_type(4)))  int          v4i;

// Pack two fp32 -> packed bf16 pair (round-to-nearest-even).
__device__ __forceinline__ unsigned int f2bf2(float lo, float hi) {
  unsigned int a = __builtin_bit_cast(unsigned int, lo);
  a += 0x7FFFu + ((a >> 16) & 1u);
  unsigned int b = __builtin_bit_cast(unsigned int, hi);
  b += 0x7FFFu + ((b >> 16) & 1u);
  return (a >> 16) | (b & 0xFFFF0000u);
}

// ---------------------------------------------------------------------------
// Async global->LDS staging (CDNA5 GLOBAL_LOAD_ASYNC_TO_LDS_B128, ASYNCcnt).
// Builtin prototype (probe-confirmed by round-2 diagnostic):
//   (int4 AS1* gsrc, int4 AS3* ldst, imm offset, imm cpol)
// Guarded by __has_builtin with a plain-staging fallback.
// ---------------------------------------------------------------------------
#if defined(__has_builtin)
#if __has_builtin(__builtin_amdgcn_global_load_async_to_lds_b128)
#define USE_ASYNC_LDS 1
#endif
#endif

#ifdef USE_ASYNC_LDS
typedef __attribute__((address_space(1))) v4i* gbl_v4i_p;
typedef __attribute__((address_space(3))) v4i* lds_v4i_p;
#endif

__device__ __forceinline__ void copy16(const float* __restrict__ g,
                                       float* __restrict__ l) {
#ifdef USE_ASYNC_LDS
  float* gg = const_cast<float*>(g);
  __builtin_amdgcn_global_load_async_to_lds_b128(
      (gbl_v4i_p)gg, (lds_v4i_p)l, /*offset=*/0, /*cpol=*/0);
#else
  *(float4*)l = *(const float4*)g;
#endif
}

__device__ __forceinline__ void tile_wait() {
#ifdef USE_ASYNC_LDS
#if __has_builtin(__builtin_amdgcn_s_wait_asynccnt)
  __builtin_amdgcn_s_wait_asynccnt(0);
#else
  asm volatile("s_wait_asynccnt 0" ::: "memory");
#endif
#endif
}

// Stage one K-slab: A tile 16x32 fp32 (2KB) + B tile 64x32 fp32 (8KB),
// B stored [n][k] so each output column's K-run is contiguous in LDS.
__device__ __forceinline__ void stage_tile(const float* __restrict__ A, int lda,
                                           const float* __restrict__ W, int ldb,
                                           int m0, int n0, int k0,
                                           float* __restrict__ la,
                                           float* __restrict__ lb, int tid) {
  {
    int m = tid >> 3, k = (tid & 7) * 4;          // 16 rows x 32 floats
    copy16(A + (size_t)(m0 + m) * lda + k0 + k, la + m * 32 + k);
  }
#pragma unroll
  for (int j = 0; j < 4; ++j) {                   // 64 rows x 32 floats
    int flat = (tid + j * 128) * 4;
    int n = flat >> 5, k = flat & 31;
    copy16(W + (size_t)(n0 + n) * ldb + k0 + k, lb + n * 32 + k);
  }
}

// A fragment (16-bit A 16x32): lane half h takes K = h*16..h*16+15 (this code's k order, same as B)
// of its row -> one contiguous 16-float run.
__device__ __forceinline__ v16bf frag_a(const float* __restrict__ ar, int kb) {
  float4 a0 = *(const float4*)(ar + kb);
  float4 a1 = *(const float4*)(ar + kb + 4);
  float4 a2 = *(const float4*)(ar + kb + 8);
  float4 a3 = *(const float4*)(ar + kb + 12);
  v8u u;
  u[0] = f2bf2(a0.x, a0.y); u[1] = f2bf2(a0.z, a0.w);
  u[2] = f2bf2(a1.x, a1.y); u[3] = f2bf2(a1.z, a1.w);
  u[4] = f2bf2(a2.x, a2.y); u[5] = f2bf2(a2.z, a2.w);
  u[6] = f2bf2(a3.x, a3.y); u[7] = f2bf2(a3.z, a3.w);
  return __builtin_bit_cast(v16bf, u);
}

// B fragment (32x16): lane half h takes K = h*16..h*16+15 of column n (same k order as A) ->
// one contiguous 16-float run in the [n][k] LDS tile.
__device__ __forceinline__ v16bf frag_b(const float* __restrict__ br) {
  float4 b0 = *(const float4*)(br);
  float4 b1 = *(const float4*)(br + 4);
  float4 b2 = *(const float4*)(br + 8);
  float4 b3 = *(const float4*)(br + 12);
  v8u u;
  u[0] = f2bf2(b0.x, b0.y); u[1] = f2bf2(b0.z, b0.w);
  u[2] = f2bf2(b1.x, b1.y); u[3] = f2bf2(b1.z, b1.w);
  u[4] = f2bf2(b2.x, b2.y); u[5] = f2bf2(b2.z, b2.w);
  u[6] = f2bf2(b3.x, b3.y); u[7] = f2bf2(b3.z, b3.w);
  return __builtin_bit_cast(v16bf, u);
}

// ---------------------------------------------------------------------------
// C[M,N] = A[M,K] @ W[N,K]^T (+ bias), fp32 in/out, bf16 WMMA compute.
// Block = 128 threads (4 wave32); each wave owns one 16x16 C tile,
// block tile = 16(M) x 64(N). grid = (N/64, M/16). Double-buffered LDS:
// async-copy slab k+1 while WMMA consumes slab k.
// ---------------------------------------------------------------------------
__global__ __launch_bounds__(128)
void gemm_bf16_wmma(const float* __restrict__ A, int lda,
                    const float* __restrict__ W, int ldb,
                    const float* __restrict__ bias,
                    float* __restrict__ C, int ldc,
                    int M, int N, int K) {
  __shared__ __align__(16) float lds_a[2][16 * 32];   //  4 KB
  __shared__ __align__(16) float lds_b[2][64 * 32];   // 16 KB

  const int tid  = threadIdx.x;
  const int wave = tid >> 5;
  const int lane = tid & 31;
  const int half = lane >> 4;      // 0: lanes 0-15, 1: lanes 16-31
  const int mr   = lane & 15;

  const int m0 = blockIdx.y * 16;
  const int n0 = blockIdx.x * 64;

  stage_tile(A, lda, W, ldb, m0, n0, 0, lds_a[0], lds_b[0], tid);

  v8f acc = {};
  int pb = 0;
  for (int k0 = 0; k0 < K; k0 += 32) {
    tile_wait();             // this wave's async copies done
    __syncthreads();         // every wave's copies done -> tile pb ready
    if (k0 + 32 < K)
      stage_tile(A, lda, W, ldb, m0, n0, k0 + 32,
                 lds_a[pb ^ 1], lds_b[pb ^ 1], tid);

    v16bf a = frag_a(&lds_a[pb][mr * 32], half * 16);
    v16bf b = frag_b(&lds_b[pb][(wave * 16 + mr) * 32 + half * 16]);
    acc = __builtin_amdgcn_wmma_f32_16x16x32_bf16(false, a, false, b,
                                                  (short)0, acc, false, false);
    pb ^= 1;
  }

  // C/D layout: lanes 0-15 -> rows m0+0..7, lanes 16-31 -> rows m0+8..15.
  const int col = n0 + wave * 16 + mr;
  float bv = (bias != nullptr && col < N) ? bias[col] : 0.0f;
#pragma unroll
  for (int r = 0; r < 8; ++r) {
    int row = m0 + half * 8 + r;
    if (row < M && col < N) C[(size_t)row * ldc + col] = acc[r] + bv;
  }
}

// ---------------------------------------------------------------------------
// Embedding gathers (fp32 rows into [step, B, E] contiguous layout).
// ---------------------------------------------------------------------------
__global__ void enc_gather_kernel(const float* __restrict__ emb,
                                  const int* __restrict__ ids,
                                  float* __restrict__ out) {
  int blk = blockIdx.x;                 // s*B + b
  int s = blk / B_, b = blk % B_;
  int id = ids[b * S_ + s];
  const float* src = emb + (size_t)id * E_;
  float* dst = out + ((size_t)s * B_ + b) * E_;
  for (int e = threadIdx.x; e < E_; e += blockDim.x) dst[e] = src[e];
}

__global__ void dec_gather_kernel(const float* __restrict__ emb,
                                  const int* __restrict__ ids,
                                  float* __restrict__ out) {
  int blk = blockIdx.x;                 // t*B + b
  int t = blk / B_, b = blk % B_;
  int id = (t == 0) ? 1 : ids[b * T_ + t];   // SOS at t=0, teacher forcing
  const float* src = emb + (size_t)id * E_;
  float* dst = out + ((size_t)t * B_ + b) * E_;
  for (int e = threadIdx.x; e < E_; e += blockDim.x) dst[e] = src[e];
}

__global__ void fill_kernel(float* __restrict__ p, int n, float v) {
  int i = blockIdx.x * blockDim.x + threadIdx.x;
  if (i < n) p[i] = v;
}

// ---------------------------------------------------------------------------
// GRU elementwise update: h' = (1-z)*n + z*h ; optionally store into enc_out.
// gi/gh are [B, 3H] with gate order [r, z, n].
// ---------------------------------------------------------------------------
__global__ void gru_update_kernel(const float* __restrict__ gi,
                                  const float* __restrict__ gh,
                                  float* __restrict__ h,
                                  float* __restrict__ store, int store_stride) {
  int i = blockIdx.x * blockDim.x + threadIdx.x;   // b*H + hh
  if (i >= B_ * H_) return;
  int b = i / H_, hh = i % H_;
  const float* gib = gi + (size_t)b * 3 * H_;
  const float* ghb = gh + (size_t)b * 3 * H_;
  float r = 1.0f / (1.0f + __expf(-(gib[hh] + ghb[hh])));
  float z = 1.0f / (1.0f + __expf(-(gib[H_ + hh] + ghb[H_ + hh])));
  float n = tanhf(gib[2 * H_ + hh] + r * ghb[2 * H_ + hh]);
  float hn = (1.0f - z) * n + z * h[i];
  h[i] = hn;
  if (store) store[(size_t)b * store_stride + hh] = hn;
}

// ---------------------------------------------------------------------------
// Bahdanau scores: scores[b,s] = tanh(q[b]+keys[b,s]) . ws + bs, masked.
// ---------------------------------------------------------------------------
__global__ __launch_bounds__(256)
void attn_scores_kernel(const float* __restrict__ q,
                        const float* __restrict__ keys,
                        const float* __restrict__ wsv,
                        const float* __restrict__ bsv,
                        const int* __restrict__ src_ids,
                        float* __restrict__ scores) {
  __shared__ float red[256];
  int blk = blockIdx.x;                 // b*S + s
  int b = blk / S_, s = blk % S_;
  const float* qb = q + (size_t)b * H_;
  const float* kb = keys + (size_t)blk * H_;
  float p = 0.0f;
  for (int hh = threadIdx.x; hh < H_; hh += 256)
    p += tanhf(qb[hh] + kb[hh]) * wsv[hh];
  red[threadIdx.x] = p;
  __syncthreads();
  for (int off = 128; off > 0; off >>= 1) {
    if ((int)threadIdx.x < off) red[threadIdx.x] += red[threadIdx.x + off];
    __syncthreads();
  }
  if (threadIdx.x == 0) {
    float sc = red[0] + bsv[0];
    if (src_ids[b * S_ + s] == 0) sc = -INFINITY;   // PAD mask
    scores[blk] = sc;
  }
}

// Softmax over S per batch row; writes attn weights to scratch and d_out.
__global__ void softmax_kernel(const float* __restrict__ scores,
                               float* __restrict__ attn_w,
                               float* __restrict__ attn_out, int t) {
  __shared__ float sm[S_];
  int b = blockIdx.x, s = threadIdx.x;   // 64 threads
  float v = scores[b * S_ + s];
  sm[s] = v; __syncthreads();
  for (int off = 32; off > 0; off >>= 1) {
    if (s < off) sm[s] = fmaxf(sm[s], sm[s + off]);
    __syncthreads();
  }
  float mx = sm[0]; __syncthreads();
  float e = __expf(v - mx);
  sm[s] = e; __syncthreads();
  for (int off = 32; off > 0; off >>= 1) {
    if (s < off) sm[s] += sm[s + off];
    __syncthreads();
  }
  float w = e / sm[0];
  attn_w[b * S_ + s] = w;
  attn_out[((size_t)b * TD + t) * S_ + s] = w;
}

// ctx[b,h] = sum_s w[b,s] * enc_out[b,s,h]
__global__ void ctx_kernel(const float* __restrict__ attn_w,
                           const float* __restrict__ enc_out,
                           float* __restrict__ ctx) {
  int i = blockIdx.x * blockDim.x + threadIdx.x;
  if (i >= B_ * H_) return;
  int b = i / H_, hh = i % H_;
  const float* w  = attn_w + b * S_;
  const float* eo = enc_out + (size_t)b * S_ * H_ + hh;
  float acc = 0.0f;
#pragma unroll 4
  for (int s = 0; s < S_; ++s) acc += w[s] * eo[(size_t)s * H_];
  ctx[i] = acc;
}

// x[b] = concat(emb_t[b], ctx[b])  -> [B, E+H]
__global__ void concat_kernel(const float* __restrict__ emb_t,
                              const float* __restrict__ ctx,
                              float* __restrict__ x) {
  int i = blockIdx.x * blockDim.x + threadIdx.x;
  if (i >= B_ * (E_ + H_)) return;
  int b = i / (E_ + H_), e = i % (E_ + H_);
  x[i] = (e < E_) ? emb_t[b * E_ + e] : ctx[b * H_ + (e - E_)];
}

// ---------------------------------------------------------------------------
extern "C" void kernel_launch(void* const* d_in, const int* in_sizes, int n_in,
                              void* d_out, int out_size, void* d_ws, size_t ws_size,
                              hipStream_t stream) {
  (void)in_sizes; (void)n_in; (void)out_size; (void)ws_size;

  const int*   src_ids   = (const int*)  d_in[0];
  const int*   tgt_ids   = (const int*)  d_in[1];
  const float* enc_embed = (const float*)d_in[2];
  const float* enc_w_ih  = (const float*)d_in[3];
  const float* enc_w_hh  = (const float*)d_in[4];
  const float* enc_b_ih  = (const float*)d_in[5];
  const float* enc_b_hh  = (const float*)d_in[6];
  const float* wq        = (const float*)d_in[7];
  const float* bq        = (const float*)d_in[8];
  const float* wk        = (const float*)d_in[9];
  const float* bk        = (const float*)d_in[10];
  const float* wsv       = (const float*)d_in[11];
  const float* bsv       = (const float*)d_in[12];
  const float* dec_embed = (const float*)d_in[13];
  const float* dec_w_ih  = (const float*)d_in[14];
  const float* dec_w_hh  = (const float*)d_in[15];
  const float* dec_b_ih  = (const float*)d_in[16];
  const float* dec_b_hh  = (const float*)d_in[17];
  const float* wo        = (const float*)d_in[18];
  const float* bo        = (const float*)d_in[19];

  float* out    = (float*)d_out;
  float* logits = out;                              // [B, TD, V]
  float* attn   = out + (size_t)B_ * TD * V_;       // [B, TD, S]

  // Workspace carve-up (~25 MB of fp32).
  float* w = (float*)d_ws;
  float* enc_emb = w; w += (size_t)S_ * B_ * E_;    // [S,B,E]
  float* dec_emb = w; w += (size_t)TD * B_ * E_;    // [TD,B,E]
  float* enc_out = w; w += (size_t)B_ * S_ * H_;    // [B,S,H]
  float* keys    = w; w += (size_t)B_ * S_ * H_;    // [B,S,H]
  float* h       = w; w += (size_t)B_ * H_;         // [B,H]
  float* gi      = w; w += (size_t)B_ * 3 * H_;     // [B,3H]
  float* gh      = w; w += (size_t)B_ * 3 * H_;     // [B,3H]
  float* q       = w; w += (size_t)B_ * H_;         // [B,H]
  float* scores  = w; w += (size_t)B_ * S_;         // [B,S]
  float* attn_w  = w; w += (size_t)B_ * S_;         // [B,S]
  float* ctx     = w; w += (size_t)B_ * H_;         // [B,H]
  float* xcat    = w; w += (size_t)B_ * (E_ + H_);  // [B,E+H]

  // ---- embeddings + h0 ----
  enc_gather_kernel<<<B_ * S_, 256, 0, stream>>>(enc_embed, src_ids, enc_emb);
  dec_gather_kernel<<<TD * B_, 256, 0, stream>>>(dec_embed, tgt_ids, dec_emb);
  fill_kernel<<<(B_ * H_ + 255) / 256, 256, 0, stream>>>(h, B_ * H_, 0.0f);

  // ---- encoder GRU scan ----
  for (int s = 0; s < S_; ++s) {
    gemm_bf16_wmma<<<dim3(3 * H_ / 64, B_ / 16), 128, 0, stream>>>(
        enc_emb + (size_t)s * B_ * E_, E_, enc_w_ih, E_, enc_b_ih,
        gi, 3 * H_, B_, 3 * H_, E_);
    gemm_bf16_wmma<<<dim3(3 * H_ / 64, B_ / 16), 128, 0, stream>>>(
        h, H_, enc_w_hh, H_, enc_b_hh, gh, 3 * H_, B_, 3 * H_, H_);
    gru_update_kernel<<<(B_ * H_ + 255) / 256, 256, 0, stream>>>(
        gi, gh, h, enc_out + (size_t)s * H_, S_ * H_);
  }

  // ---- keys_proj = enc_out @ wk^T + bk (one big GEMM, hoisted) ----
  gemm_bf16_wmma<<<dim3(H_ / 64, (B_ * S_) / 16), 128, 0, stream>>>(
      enc_out, H_, wk, H_, bk, keys, H_, B_ * S_, H_, H_);

  // ---- decoder scan with attention + teacher forcing ----
  for (int t = 0; t < TD; ++t) {
    gemm_bf16_wmma<<<dim3(H_ / 64, B_ / 16), 128, 0, stream>>>(
        h, H_, wq, H_, bq, q, H_, B_, H_, H_);
    attn_scores_kernel<<<B_ * S_, 256, 0, stream>>>(q, keys, wsv, bsv,
                                                    src_ids, scores);
    softmax_kernel<<<B_, S_, 0, stream>>>(scores, attn_w, attn, t);
    ctx_kernel<<<(B_ * H_ + 255) / 256, 256, 0, stream>>>(attn_w, enc_out, ctx);
    concat_kernel<<<(B_ * (E_ + H_) + 255) / 256, 256, 0, stream>>>(
        dec_emb + (size_t)t * B_ * E_, ctx, xcat);
    gemm_bf16_wmma<<<dim3(3 * H_ / 64, B_ / 16), 128, 0, stream>>>(
        xcat, E_ + H_, dec_w_ih, E_ + H_, dec_b_ih, gi, 3 * H_, B_, 3 * H_, E_ + H_);
    gemm_bf16_wmma<<<dim3(3 * H_ / 64, B_ / 16), 128, 0, stream>>>(
        h, H_, dec_w_hh, H_, dec_b_hh, gh, 3 * H_, B_, 3 * H_, H_);
    gru_update_kernel<<<(B_ * H_ + 255) / 256, 256, 0, stream>>>(
        gi, gh, h, nullptr, 0);
    // logits GEMM dominates: [32 x 32000 x 1024]; wo stays resident in L2.
    gemm_bf16_wmma<<<dim3(V_ / 64, B_ / 16), 128, 0, stream>>>(
        h, H_, wo, H_, bo, logits + (size_t)t * V_, TD * V_, B_, V_, H_);
  }
}
